// HGT_5592047419502
// MI455X (gfx1250) — matrix operand order repaired and checked by reading, not compiled, not run on hardware
//
#include <hip/hip_runtime.h>
#include <math.h>

// ---------------------------------------------------------------------------
// HGT forward for MI455X (gfx1250, wave32, WMMA).
// Dense node GEMMs via v_wmma_f32_16x16x32_f16 with b128 LDS fragment loads;
// edge softmax via atomics.
// ---------------------------------------------------------------------------

typedef __attribute__((ext_vector_type(16))) _Float16 v16h;
typedef __attribute__((ext_vector_type(8)))  _Float16 v8h;
typedef __attribute__((ext_vector_type(8)))  float    v8f;

#define NT_  500000
#define NC_  500000
#define ETT_ 1000000
#define ECT_ 1000000

// ---------------- WMMA fragment helpers (wave32 layouts, ISA 7.12.2) -------

// A fragment: 16x32 f16 tile from a 16x64 row-major f16 LDS buffer, cols k0..k0+31.
// lanes 0-15 : row M=lane,    elems 0-7 -> K=k0+e,    elems 8-15 -> K=k0+16+(e-8)
// lanes 16-31: row M=lane-16, elems 0-7 -> K=k0+8+e,  elems 8-15 -> K=k0+24+(e-8)
// Both runs are 16B-contiguous -> two ds_load_b128 per fragment.
__device__ __forceinline__ v16h frag_a(const _Float16* s, int lane, int k0) {
    int row = lane & 15;
    int khi = (lane >> 4) & 1;
    const _Float16* base = s + row * 64 + k0 + khi * 8;
    v8h lo = *(const v8h*)(base);        // K = k0+khi*8   .. +7
    v8h hi = *(const v8h*)(base + 16);   // K = k0+16+khi*8 .. +7
    return __builtin_shufflevector(lo, hi, 0, 1, 2, 3, 4, 5, 6, 7,
                                           8, 9, 10, 11, 12, 13, 14, 15);
}

// B fragment: 32x16 f16 tile; weight staged TRANSPOSED in LDS as sWt[n*64 + k]
// so each lane's 16 K-values (N fixed) are two 16B-contiguous runs (K=kb..+7, kb+16..+23) -> 2x ds_load_b128.
__device__ __forceinline__ v16h frag_b(const _Float16* wt, int lane, int k0, int n0) {
    int n  = (lane & 15) + n0;
    int kb = k0 + ((lane >> 4) & 1) * 8;
    const _Float16* base = wt + n * 64 + kb;
    v8h lo = *(const v8h*)(base);        // K = kb .. kb+7
    v8h hi = *(const v8h*)(base + 16);   // K = kb+16 .. kb+23
    return __builtin_shufflevector(lo, hi, 0, 1, 2, 3, 4, 5, 6, 7,
                                           8, 9, 10, 11, 12, 13, 14, 15);
}

// Full 16x64 @ 64x64 tile GEMM: 8 WMMA ops, fp32 accumulate.
__device__ __forceinline__ void gemm_tile(const _Float16* sIn, const _Float16* sWt,
                                          int lane, v8f acc[4]) {
    v16h a0 = frag_a(sIn, lane, 0);
    v16h a1 = frag_a(sIn, lane, 32);
#pragma unroll
    for (int nt = 0; nt < 4; ++nt) {
        v8f c = {0.f, 0.f, 0.f, 0.f, 0.f, 0.f, 0.f, 0.f};
        v16h b0 = frag_b(sWt, lane, 0,  nt * 16);
        v16h b1 = frag_b(sWt, lane, 32, nt * 16);
        c = __builtin_amdgcn_wmma_f32_16x16x32_f16(false, a0, false, b0, (short)0, c, false, false);
        c = __builtin_amdgcn_wmma_f32_16x16x32_f16(false, a1, false, b1, (short)0, c, false, false);
        acc[nt] = c;
    }
}

// D layout: vgpr r -> M = r + 8*(lane>=16); N = (lane&15) + 16*nt.
__device__ __forceinline__ void store_lds(const v8f acc[4], const float* bias,
                                          _Float16* sOut, int lane, bool relu) {
    int nb = lane & 15;
    int mb = ((lane >> 4) & 1) * 8;
#pragma unroll
    for (int nt = 0; nt < 4; ++nt) {
        int n = nt * 16 + nb;
        float bv = bias ? bias[n] : 0.f;
#pragma unroll
        for (int r = 0; r < 8; ++r) {
            float x = acc[nt][r] + bv;
            if (relu) x = fmaxf(x, 0.f);
            sOut[(mb + r) * 64 + n] = (_Float16)x;
        }
    }
    asm volatile("s_wait_dscnt 0" ::: "memory"); // cross-lane LDS RAW fence
}

__device__ __forceinline__ void store_global(const v8f acc[4], const float* bias,
                                             float* out, size_t m0, int lane, bool relu) {
    int nb = lane & 15;
    int mb = ((lane >> 4) & 1) * 8;
#pragma unroll
    for (int nt = 0; nt < 4; ++nt) {
        int n = nt * 16 + nb;
        float bv = bias ? bias[n] : 0.f;
#pragma unroll
        for (int r = 0; r < 8; ++r) {
            float x = acc[nt][r] + bv;
            if (relu) x = fmaxf(x, 0.f);
            out[(m0 + mb + r) * 64 + n] = x;
        }
    }
}

// Stage a 16-row fp32 input tile (ncols valid, multiple of 8, zero-padded to 64)
// as f16 with 16B LDS stores.
__device__ __forceinline__ void stage_in_f32(const float* x, size_t m0, int ld, int ncols,
                                             _Float16* sIn, int lane) {
    for (int i = lane; i < 16 * 8; i += 32) {
        int r = i >> 3, c8 = (i & 7) * 8;
        v8h h = (v8h)(_Float16)0.f;
        if (c8 < ncols) {
            const float4* p = (const float4*)(x + (m0 + r) * (size_t)ld + c8);
            float4 a = p[0], b = p[1];
            h[0] = (_Float16)a.x; h[1] = (_Float16)a.y;
            h[2] = (_Float16)a.z; h[3] = (_Float16)a.w;
            h[4] = (_Float16)b.x; h[5] = (_Float16)b.y;
            h[6] = (_Float16)b.z; h[7] = (_Float16)b.w;
        }
        *(v8h*)(sIn + r * 64 + c8) = h;
    }
    asm volatile("s_wait_dscnt 0" ::: "memory");
}

// Stage a [rows x 64] fp32 weight TRANSPOSED into LDS f16: sWt[n*64+k] = W[k*64+n].
__device__ __forceinline__ void stage_wT(const float* W, _Float16* sWt,
                                         int tid, int nthr, int rows) {
    for (int i = tid; i < 64 * 64; i += nthr) {
        int n = i >> 6, k = i & 63;
        sWt[i] = (_Float16)((k < rows) ? W[k * 64 + n] : 0.f);
    }
}

// Stage per-head [4][16][16] relation tensor as TRANSPOSED block-diagonal 64x64 f16:
// sRt[n*64+k] = (n>>4 == k>>4) ? A[h][k&15][n&15] : 0
__device__ __forceinline__ void stage_blockdiagT(const float* A, _Float16* sRt,
                                                 int tid, int nthr) {
    for (int i = tid; i < 64 * 64; i += nthr) {
        int n = i >> 6, k = i & 63;
        int h = n >> 4;
        float v = ((k >> 4) == h) ? A[h * 256 + (k & 15) * 16 + (n & 15)] : 0.f;
        sRt[i] = (_Float16)v;
    }
}

// ---------------- node GEMM kernels ----------------------------------------

// 3-layer MLP: h = relu(relu(relu(x@w1+b1)@w2+b2)@w3+b3)
__global__ void __launch_bounds__(128)
k_mlp(const float* __restrict__ x, int ldx, int ncols,
      const float* __restrict__ w1, int rows1, const float* __restrict__ b1,
      const float* __restrict__ w2, const float* __restrict__ b2,
      const float* __restrict__ w3, const float* __restrict__ b3,
      float* __restrict__ hout, int n) {
    __shared__ alignas(16) _Float16 sW1[4096], sW2[4096], sW3[4096];
    __shared__ alignas(16) _Float16 sBuf[4][2][16 * 64];
    int tid = threadIdx.x, lane = tid & 31, wv = tid >> 5;
    stage_wT(w1, sW1, tid, 128, rows1);
    stage_wT(w2, sW2, tid, 128, 64);
    stage_wT(w3, sW3, tid, 128, 64);
    __syncthreads();
    size_t m0 = ((size_t)blockIdx.x * 4 + wv) * 16;
    if (m0 >= (size_t)n) return;
    _Float16* sIn = sBuf[wv][0];
    _Float16* sOut = sBuf[wv][1];
    v8f acc[4];
    stage_in_f32(x, m0, ldx, ncols, sIn, lane);
    gemm_tile(sIn, sW1, lane, acc);  store_lds(acc, b1, sOut, lane, true);
    gemm_tile(sOut, sW2, lane, acc); store_lds(acc, b2, sIn, lane, true);
    gemm_tile(sIn, sW3, lane, acc);  store_global(acc, b3, hout, m0, lane, true);
}

// K/Q/V projections + block-diag relation transforms.
// q = h@qw+qb (optional); kr = (h@kw+kb)@Ra; vr = (h@vw+vb)@Rm
__global__ void __launch_bounds__(128)
k_kqvr(const float* __restrict__ hin,
       const float* __restrict__ kw, const float* __restrict__ kb,
       const float* __restrict__ qw, const float* __restrict__ qb,
       const float* __restrict__ vw, const float* __restrict__ vb,
       const float* __restrict__ arel, const float* __restrict__ mrel,
       float* __restrict__ qout, float* __restrict__ krout, float* __restrict__ vrout,
       int n, int has_q) {
    __shared__ alignas(16) _Float16 sKW[4096], sQW[4096], sVW[4096], sRA[4096], sRM[4096];
    __shared__ alignas(16) _Float16 sBuf[4][2][16 * 64];
    int tid = threadIdx.x, lane = tid & 31, wv = tid >> 5;
    stage_wT(kw, sKW, tid, 128, 64);
    if (has_q) stage_wT(qw, sQW, tid, 128, 64);
    stage_wT(vw, sVW, tid, 128, 64);
    stage_blockdiagT(arel, sRA, tid, 128);
    stage_blockdiagT(mrel, sRM, tid, 128);
    __syncthreads();
    size_t m0 = ((size_t)blockIdx.x * 4 + wv) * 16;
    if (m0 >= (size_t)n) return;
    _Float16* sIn = sBuf[wv][0];
    _Float16* sTmp = sBuf[wv][1];
    v8f acc[4];
    stage_in_f32(hin, m0, 64, 64, sIn, lane);
    if (has_q) {
        gemm_tile(sIn, sQW, lane, acc);
        store_global(acc, qb, qout, m0, lane, false);
    }
    gemm_tile(sIn, sKW, lane, acc);  store_lds(acc, kb, sTmp, lane, false);
    gemm_tile(sTmp, sRA, lane, acc); store_global(acc, nullptr, krout, m0, lane, false);
    gemm_tile(sIn, sVW, lane, acc);  store_lds(acc, vb, sTmp, lane, false);
    gemm_tile(sTmp, sRM, lane, acc); store_global(acc, nullptr, vrout, m0, lane, false);
}

// out = sigmoid(skip) * (gelu(agg)@aw + ab) + (1-sigmoid(skip)) * ht
__global__ void __launch_bounds__(128)
k_final(const float* __restrict__ agg, const float* __restrict__ ht,
        const float* __restrict__ aw, const float* __restrict__ ab,
        const float* __restrict__ skip, float* __restrict__ out, int n) {
    __shared__ alignas(16) _Float16 sAW[4096];
    __shared__ alignas(16) _Float16 sIn[4][16 * 64];
    int tid = threadIdx.x, lane = tid & 31, wv = tid >> 5;
    stage_wT(aw, sAW, tid, 128, 64);
    __syncthreads();
    size_t m0 = ((size_t)blockIdx.x * 4 + wv) * 16;
    if (m0 >= (size_t)n) return;
    float g = 1.f / (1.f + __expf(-skip[0]));
    // stage gelu(agg) tile as f16 with 16B LDS stores
    for (int i = lane; i < 16 * 8; i += 32) {
        int r = i >> 3, c8 = (i & 7) * 8;
        const float4* p = (const float4*)(agg + (m0 + r) * 64 + c8);
        float4 a = p[0], b = p[1];
        float xs[8] = {a.x, a.y, a.z, a.w, b.x, b.y, b.z, b.w};
        v8h h;
#pragma unroll
        for (int j = 0; j < 8; ++j) {
            float xv = xs[j];
            h[j] = (_Float16)(0.5f * xv * (1.f + erff(xv * 0.70710678118654752f)));
        }
        *(v8h*)(sIn[wv] + r * 64 + c8) = h;
    }
    asm volatile("s_wait_dscnt 0" ::: "memory");
    v8f acc[4];
    gemm_tile(sIn[wv], sAW, lane, acc);
    int nb = lane & 15;
    int mb = ((lane >> 4) & 1) * 8;
#pragma unroll
    for (int nt = 0; nt < 4; ++nt) {
        int nn = nt * 16 + nb;
        float bv = ab[nn];
#pragma unroll
        for (int r = 0; r < 8; ++r) {
            size_t row = m0 + mb + r;
            float v = acc[nt][r] + bv;
            out[row * 64 + nn] = g * v + (1.f - g) * ht[row * 64 + nn];
        }
    }
}

// ---------------- edge-softmax kernels -------------------------------------

// order-preserving float -> uint encoding for atomic max
__device__ __forceinline__ unsigned enc_f(float f) {
    unsigned u = __float_as_uint(f);
    return (u & 0x80000000u) ? ~u : (u | 0x80000000u);
}
__device__ __forceinline__ float dec_f(unsigned u) {
    return (u & 0x80000000u) ? __uint_as_float(u & 0x7fffffffu) : __uint_as_float(~u);
}

__global__ void k_init(unsigned* __restrict__ mEnc, float* __restrict__ den, size_t n4) {
    size_t i = (size_t)blockIdx.x * blockDim.x + threadIdx.x;
    if (i >= n4) return;
    mEnc[i] = 0x007FFFFFu; // enc(-inf)
    den[i] = 0.f;
}

__global__ void k_zero(float* __restrict__ p, size_t n) {
    size_t i = (size_t)blockIdx.x * blockDim.x + threadIdx.x;
    if (i < n) p[i] = 0.f;
}

// pass 1: raw scores + running per-(dst,head) max
__global__ void edge_score_max(const int* __restrict__ ei, int E,
                               const float* __restrict__ q, const float* __restrict__ kr,
                               const float* __restrict__ p,
                               float* __restrict__ score, unsigned* __restrict__ mEnc) {
    int e = blockIdx.x * blockDim.x + threadIdx.x;
    if (e >= E) return;
    int s = ei[e], d = ei[E + e];
    const float4* qv = (const float4*)(q + (size_t)d * 64);
    const float4* kv = (const float4*)(kr + (size_t)s * 64);
    __builtin_prefetch((const void*)kv, 0, 0); // global_prefetch_b8
#pragma unroll
    for (int h = 0; h < 4; ++h) {
        float acc = 0.f;
#pragma unroll
        for (int j = 0; j < 4; ++j) {
            float4 a = qv[h * 4 + j], b = kv[h * 4 + j];
            acc += a.x * b.x + a.y * b.y + a.z * b.z + a.w * b.w;
        }
        float sc = acc * p[h] * 0.25f; // scale = 1/sqrt(D), D = 16
        score[(size_t)e * 4 + h] = sc;
        atomicMax(&mEnc[(size_t)d * 4 + h], enc_f(sc));
    }
}

// pass 2: e = exp(s - m); den[dst,h] += e; overwrite score with e
__global__ void edge_exp_den(const int* __restrict__ ei, int E,
                             const unsigned* __restrict__ mEnc,
                             float* __restrict__ score, float* __restrict__ den) {
    int e = blockIdx.x * blockDim.x + threadIdx.x;
    if (e >= E) return;
    int d = ei[E + e];
#pragma unroll
    for (int h = 0; h < 4; ++h) {
        float m = dec_f(mEnc[(size_t)d * 4 + h]);
        if (!isfinite(m)) m = 0.f;
        float ex = expf(score[(size_t)e * 4 + h] - m);
        score[(size_t)e * 4 + h] = ex;
        atomicAdd(&den[(size_t)d * 4 + h], ex);
    }
}

// pass 3: one thread per (edge, head): agg[dst, h*16+j] += alpha * vr[src, h*16+j]
__global__ void edge_agg(const int* __restrict__ ei, int E,
                         const float* __restrict__ score, const float* __restrict__ den,
                         const float* __restrict__ vr, float* __restrict__ agg) {
    size_t t = (size_t)blockIdx.x * blockDim.x + threadIdx.x;
    if (t >= (size_t)E * 4) return;
    int e = (int)(t >> 2), h = (int)(t & 3);
    int s = ei[e], d = ei[E + e];
    float alpha = score[(size_t)e * 4 + h] / fmaxf(den[(size_t)d * 4 + h], 1e-16f);
    const float* v = vr + (size_t)s * 64 + h * 16;
    float* o = agg + (size_t)d * 64 + h * 16;
#pragma unroll
    for (int j = 0; j < 16; ++j) atomicAdd(&o[j], alpha * v[j]);
}

// ---------------- launcher -------------------------------------------------

extern "C" void kernel_launch(void* const* d_in, const int* in_sizes, int n_in,
                              void* d_out, int out_size, void* d_ws, size_t ws_size,
                              hipStream_t stream) {
    (void)in_sizes; (void)n_in; (void)out_size; (void)ws_size;
    const float* x_t  = (const float*)d_in[0];
    const float* x_c  = (const float*)d_in[1];
    const int*   eitt = (const int*)d_in[2];
    const int*   eict = (const int*)d_in[3];
    const float *wt1 = (const float*)d_in[4],  *bt1 = (const float*)d_in[5];
    const float *wt2 = (const float*)d_in[6],  *bt2 = (const float*)d_in[7];
    const float *wt3 = (const float*)d_in[8],  *bt3 = (const float*)d_in[9];
    const float *wc1 = (const float*)d_in[10], *bc1 = (const float*)d_in[11];
    const float *wc2 = (const float*)d_in[12], *bc2 = (const float*)d_in[13];
    const float *wc3 = (const float*)d_in[14], *bc3 = (const float*)d_in[15];
    const float *kw_t = (const float*)d_in[16], *kb_t = (const float*)d_in[17];
    const float *qw_t = (const float*)d_in[18], *qb_t = (const float*)d_in[19];
    const float *vw_t = (const float*)d_in[20], *vb_t = (const float*)d_in[21];
    const float *kw_c = (const float*)d_in[22], *kb_c = (const float*)d_in[23];
    const float *vw_c = (const float*)d_in[24], *vb_c = (const float*)d_in[25];
    const float *a_tt = (const float*)d_in[26], *m_tt = (const float*)d_in[27];
    const float *p_tt = (const float*)d_in[28];
    const float *a_ct = (const float*)d_in[29], *m_ct = (const float*)d_in[30];
    const float *p_ct = (const float*)d_in[31];
    const float *aw = (const float*)d_in[32], *ab = (const float*)d_in[33];
    const float *skip = (const float*)d_in[34];

    float* ws = (float*)d_ws;
    size_t o = 0;
    float* ht   = ws + o; o += (size_t)NT_ * 64;
    float* qt   = ws + o; o += (size_t)NT_ * 64;   // reused as agg after pass 1
    float* krtt = ws + o; o += (size_t)NT_ * 64;
    float* vrtt = ws + o; o += (size_t)NT_ * 64;
    float* krct = ws + o; o += (size_t)NC_ * 64;
    float* vrct = ws + o; o += (size_t)NC_ * 64;
    float* hc   = ws + o; o += (size_t)NC_ * 64;
    unsigned* mEnc = (unsigned*)(ws + o); o += (size_t)NT_ * 4;
    float* den  = ws + o; o += (size_t)NT_ * 4;
    float* score = ws + o; o += (size_t)(ETT_ + ECT_) * 4;
    float* agg = qt;

    dim3 b128(128), b256(256);
    int gT = (NT_ + 63) / 64;
    int gC = (NC_ + 63) / 64;

    k_init<<<(NT_ * 4 + 255) / 256, b256, 0, stream>>>(mEnc, den, (size_t)NT_ * 4);

    k_mlp<<<gT, b128, 0, stream>>>(x_t, 64, 64, wt1, 64, bt1, wt2, bt2, wt3, bt3, ht, NT_);
    k_mlp<<<gC, b128, 0, stream>>>(x_c, 48, 48, wc1, 48, bc1, wc2, bc2, wc3, bc3, hc, NC_);

    k_kqvr<<<gT, b128, 0, stream>>>(ht, kw_t, kb_t, qw_t, qb_t, vw_t, vb_t,
                                    a_tt, m_tt, qt, krtt, vrtt, NT_, 1);
    k_kqvr<<<gC, b128, 0, stream>>>(hc, kw_c, kb_c, kw_c, kb_c, vw_c, vb_c,
                                    a_ct, m_ct, nullptr, krct, vrct, NC_, 0);

    edge_score_max<<<(ETT_ + 255) / 256, b256, 0, stream>>>(eitt, ETT_, qt, krtt, p_tt,
                                                            score, mEnc);
    edge_score_max<<<(ECT_ + 255) / 256, b256, 0, stream>>>(eict, ECT_, qt, krct, p_ct,
                                                            score + (size_t)ETT_ * 4, mEnc);

    // q_t is dead now; zero its buffer for use as agg
    k_zero<<<(int)(((size_t)NT_ * 64 + 255) / 256), b256, 0, stream>>>(agg, (size_t)NT_ * 64);

    edge_exp_den<<<(ETT_ + 255) / 256, b256, 0, stream>>>(eitt, ETT_, mEnc, score, den);
    edge_exp_den<<<(ECT_ + 255) / 256, b256, 0, stream>>>(eict, ECT_, mEnc,
                                                          score + (size_t)ETT_ * 4, den);

    edge_agg<<<(int)(((size_t)ETT_ * 4 + 255) / 256), b256, 0, stream>>>(
        eitt, ETT_, score, den, vrtt, agg);
    edge_agg<<<(int)(((size_t)ECT_ * 4 + 255) / 256), b256, 0, stream>>>(
        eict, ECT_, score + (size_t)ETT_ * 4, den, vrct, agg);

    k_final<<<gT, b128, 0, stream>>>(agg, ht, aw, ab, skip, (float*)d_out, NT_);
}
